// VanillaConvolutionalLayer_4836133175447
// MI455X (gfx1250) — matrix operand order repaired and checked by reading, not compiled, not run on hardware
//
#include <hip/hip_runtime.h>

// GNN message-passing layer for MI455X (gfx1250, wave32, WMMA).
// Edge MLP [600k,288]x[288,32] and node MLP [50k,160]x[160,128] run on
// v_wmma_f32_16x16x32_bf16 with f32 accumulation. Node-row gathers use the
// CDNA5 async memory->LDS path (global_load_async_to_lds_b128, ASYNCcnt),
// bypassing the VGPR round-trip; message scatter uses non-returning
// global_atomic_add_f32.

#define N_NODES 50000
#define N_EDGES 600000
#define F_NODE 128
#define F_EDGE 32
#define MSG_DIM 32
#define EDGE_IN 288  // 2*F_NODE + F_EDGE
#define NODE_IN 160  // F_NODE + MSG_DIM

typedef __bf16 v16bf __attribute__((ext_vector_type(16)));
typedef float v8f __attribute__((ext_vector_type(8)));

__device__ __forceinline__ unsigned short f2bf(float f) {
  unsigned int u = __float_as_uint(f);
  u += 0x7FFFu + ((u >> 16) & 1u);  // round-to-nearest-even
  return (unsigned short)(u >> 16);
}

// Generic pointer -> workgroup-relative LDS byte offset (flat LDS aperture
// truncates to addr[31:0] per the CDNA5 aperture rules).
__device__ __forceinline__ unsigned lds_off(const void* p) {
  return (unsigned)(uintptr_t)p;
}

// Async copy 16B global -> LDS, tracked on ASYNCcnt.
__device__ __forceinline__ void async_g2l_b128(unsigned ldsByteOff,
                                               const void* gaddr) {
  asm volatile("global_load_async_to_lds_b128 %0, %1, off"
               :
               : "v"(ldsByteOff), "v"((unsigned long long)(uintptr_t)gaddr)
               : "memory");
}

__device__ __forceinline__ void wait_asynccnt0() {
  asm volatile("s_wait_asynccnt 0x0" ::: "memory");
}

// Load a 16-halfword WMMA fragment as two 16B chunks: p[0..7] and p[gap..gap+7].
// A and B fragments both use gap=16 (K and K+16 chunks).
__device__ __forceinline__ v16bf load_frag(const unsigned short* p, int gapElems) {
  union { uint4 u[2]; v16bf v; } t;
  t.u[0] = *(const uint4*)(p);
  t.u[1] = *(const uint4*)(p + gapElems);
  return t.v;
}

// ---------------- prep: bf16 node table, transposed bf16 weights, zero msg ----
__global__ void prep_kernel(const float* __restrict__ nf,
                            const float* __restrict__ We,
                            const float* __restrict__ Wn,
                            unsigned short* __restrict__ nf16,
                            float* __restrict__ msg,
                            unsigned short* __restrict__ WeT,
                            unsigned short* __restrict__ WnT) {
  int i = blockIdx.x * blockDim.x + threadIdx.x;
  if (i < N_NODES * F_NODE) nf16[i] = f2bf(nf[i]);
  if (i < N_NODES * MSG_DIM) msg[i] = 0.0f;
  if (i < EDGE_IN * MSG_DIM) {           // We[k][n] -> WeT[n][k], bf16
    int k = i / MSG_DIM, n = i % MSG_DIM;
    WeT[n * EDGE_IN + k] = f2bf(We[i]);
  }
  if (i < NODE_IN * F_NODE) {            // Wn[k][n] -> WnT[n][k], bf16
    int k = i / F_NODE, n = i % F_NODE;
    WnT[n * NODE_IN + k] = f2bf(Wn[i]);
  }
}

// ---------------- edge MLP + scatter: one wave per 16-edge tile ---------------
#define A_STRIDE_E 304  // 288 padded; x2B = 608B row stride, 16B aligned

__global__ void __launch_bounds__(128) edge_kernel(
    const unsigned short* __restrict__ nf16,
    const int* __restrict__ eidx,       // [2][N_EDGES]
    const float* __restrict__ ef,
    const unsigned short* __restrict__ WeT,  // [32][288] bf16
    const float* __restrict__ be,
    float* __restrict__ msg) {
  __shared__ unsigned short aT[4][16 * A_STRIDE_E];  // 38,912 B
  const int wave = threadIdx.x >> 5;
  const int lane = threadIdx.x & 31;
  const int n    = lane & 15;
  const int hi   = lane >> 4;
  const int tile = blockIdx.x * 4 + wave;            // 37500 tiles exactly
  const int eBase = tile * 16;

  // Stage A tile rows: [nf16[node0] | nf16[node1] | bf16(edge_feat)] (16 x 288)
  // Node rows go memory->LDS via async b128: lanes 0-15 carry node0's row,
  // lanes 16-31 carry node1's row (16 lanes x 8 bf16 = 128 elements).
  unsigned short* a = aT[wave];
  for (int r = 0; r < 16; ++r) {
    int e = eBase + r;
    int nsel = hi ? eidx[N_EDGES + e] : eidx[e];
    unsigned short* dst = a + r * A_STRIDE_E;
    async_g2l_b128(lds_off(dst + hi * F_NODE + n * 8),
                   nf16 + (size_t)nsel * F_NODE + n * 8);
    if (lane < 8) {  // edge feats need f32->bf16 conversion: VALU path
      float4 v = ((const float4*)(ef + (size_t)e * F_EDGE))[lane];
      uint2 p;
      p.x = (unsigned)f2bf(v.x) | ((unsigned)f2bf(v.y) << 16);
      p.y = (unsigned)f2bf(v.z) | ((unsigned)f2bf(v.w) << 16);
      ((uint2*)(dst + 2 * F_NODE))[lane] = p;
    }
  }
  wait_asynccnt0();  // drain async LDS writes
  __syncthreads();   // cross-lane LDS visibility (uniform: every tile is valid)

  v8f c0 = {0.f, 0.f, 0.f, 0.f, 0.f, 0.f, 0.f, 0.f};
  v8f c1 = c0;
  const unsigned short* arow  = a + n * A_STRIDE_E + hi * 8;
  const unsigned short* b0row = WeT + (size_t)n * EDGE_IN + hi * 8;
  const unsigned short* b1row = WeT + (size_t)(16 + n) * EDGE_IN + hi * 8;
#pragma unroll
  for (int s = 0; s < 9; ++s) {  // K = 288 = 9 * 32
    v16bf af  = load_frag(arow + 32 * s, 16);
    v16bf bf0 = load_frag(b0row + 32 * s, 16);
    v16bf bf1 = load_frag(b1row + 32 * s, 16);
    c0 = __builtin_amdgcn_wmma_f32_16x16x32_bf16(false, af, false, bf0,
                                                 (short)0, c0, false, false);
    c1 = __builtin_amdgcn_wmma_f32_16x16x32_bf16(false, af, false, bf1,
                                                 (short)0, c1, false, false);
  }

  // Bias + ReLU + scatter-add into msg_sums[node0] (non-returning f32 atomics).
  float bias0 = be[n], bias1 = be[16 + n];
#pragma unroll
  for (int v = 0; v < 8; ++v) {
    int m = v + hi * 8;                    // local edge row (C/D layout)
    int n0 = eidx[eBase + m];
    float x0 = fmaxf(c0[v] + bias0, 0.0f);
    float x1 = fmaxf(c1[v] + bias1, 0.0f);
    __hip_atomic_fetch_add(msg + (size_t)n0 * MSG_DIM + n, x0,
                           __ATOMIC_RELAXED, __HIP_MEMORY_SCOPE_AGENT);
    __hip_atomic_fetch_add(msg + (size_t)n0 * MSG_DIM + 16 + n, x1,
                           __ATOMIC_RELAXED, __HIP_MEMORY_SCOPE_AGENT);
  }
}

// ---------------- node MLP: one wave per 16-node tile -------------------------
#define A_STRIDE_N 176  // 160 padded; x2B = 352B row stride, 16B aligned
#define N_TILES 3125    // 50000 / 16

__global__ void __launch_bounds__(128) node_kernel(
    const unsigned short* __restrict__ nf16,
    const float* __restrict__ msg,
    const unsigned short* __restrict__ WnT,  // [128][160] bf16
    const float* __restrict__ bn,
    float* __restrict__ out) {
  __shared__ unsigned short aT[4][16 * A_STRIDE_N];  // 22,528 B
  const int wave = threadIdx.x >> 5;
  const int lane = threadIdx.x & 31;
  const int n    = lane & 15;
  const int hi   = lane >> 4;
  const int tile = blockIdx.x * 4 + wave;
  const bool valid = tile < N_TILES;
  const int nodeBase = tile * 16;
  unsigned short* a = aT[wave];

  if (valid) {
    // Node-feature halves of two rows per async issue (16 lanes per row).
    for (int rr = 0; rr < 16; rr += 2) {
      int row = rr + hi;
      int node = nodeBase + row;
      async_g2l_b128(lds_off(a + row * A_STRIDE_N + n * 8),
                     nf16 + (size_t)node * F_NODE + n * 8);
    }
    // msg_sums need f32->bf16 conversion: VALU path.
    for (int r = 0; r < 16; ++r) {
      if (lane < 8) {
        int node = nodeBase + r;
        float4 v = ((const float4*)(msg + (size_t)node * MSG_DIM))[lane];
        uint2 p;
        p.x = (unsigned)f2bf(v.x) | ((unsigned)f2bf(v.y) << 16);
        p.y = (unsigned)f2bf(v.z) | ((unsigned)f2bf(v.w) << 16);
        ((uint2*)(a + r * A_STRIDE_N + F_NODE))[lane] = p;
      }
    }
    wait_asynccnt0();
  }
  __syncthreads();
  if (!valid) return;

  v8f z = {0.f, 0.f, 0.f, 0.f, 0.f, 0.f, 0.f, 0.f};
  v8f c[8];
#pragma unroll
  for (int t = 0; t < 8; ++t) c[t] = z;

  const unsigned short* arow = a + n * A_STRIDE_N + hi * 8;
#pragma unroll
  for (int s = 0; s < 5; ++s) {  // K = 160 = 5 * 32
    v16bf af = load_frag(arow + 32 * s, 16);
#pragma unroll
    for (int t = 0; t < 8; ++t) {  // N = 128 = 8 * 16
      const unsigned short* brow =
          WnT + (size_t)(t * 16 + n) * NODE_IN + 32 * s + hi * 8;
      v16bf bfb = load_frag(brow, 16);
      c[t] = __builtin_amdgcn_wmma_f32_16x16x32_bf16(false, af, false, bfb,
                                                     (short)0, c[t], false, false);
    }
  }

#pragma unroll
  for (int t = 0; t < 8; ++t) {
    float bias = bn[t * 16 + n];
#pragma unroll
    for (int v = 0; v < 8; ++v) {
      int m = v + hi * 8;
      float x = fmaxf(c[t][v] + bias, 0.0f);
      out[(size_t)(nodeBase + m) * F_NODE + t * 16 + n] = x;
    }
  }
}

// ---------------- host launcher ----------------------------------------------
extern "C" void kernel_launch(void* const* d_in, const int* in_sizes, int n_in,
                              void* d_out, int out_size, void* d_ws, size_t ws_size,
                              hipStream_t stream) {
  const float* nf  = (const float*)d_in[0];
  const int* eidx  = (const int*)d_in[1];   // edge_node_indices [2][E]
  const float* ef  = (const float*)d_in[2];
  const float* We  = (const float*)d_in[3];
  const float* be  = (const float*)d_in[4];
  const float* Wn  = (const float*)d_in[5];
  const float* bn  = (const float*)d_in[6];
  float* out = (float*)d_out;

  // Workspace layout (~19.3 MB total):
  char* ws = (char*)d_ws;
  unsigned short* nf16 = (unsigned short*)ws;               // 12,800,000 B
  float* msg           = (float*)(ws + 12800000);           //  6,400,000 B
  unsigned short* WeT  = (unsigned short*)(ws + 19200000);  //     18,432 B
  unsigned short* WnT  = (unsigned short*)(ws + 19218432);  //     40,960 B

  // 1) bf16 node table + transposed bf16 weights + zero the f32 accumulator
  //    (must re-zero every call: harness does not reset workspace).
  int prepN = N_NODES * F_NODE;  // 6.4M covers all prep tasks
  prep_kernel<<<(prepN + 255) / 256, 256, 0, stream>>>(nf, We, Wn, nf16, msg,
                                                       WeT, WnT);

  // 2) edge MLP + atomic scatter: 37500 tiles, 4 waves/block -> 9375 blocks
  edge_kernel<<<N_EDGES / 16 / 4, 128, 0, stream>>>(nf16, eidx, ef, WeT, be, msg);

  // 3) node MLP: 3125 tiles, 4 waves/block -> 782 blocks (tail guarded)
  node_kernel<<<(N_TILES + 3) / 4, 128, 0, stream>>>(nf16, msg, WnT, bn, out);
}
